// JointVGAE_54030688584368
// MI455X (gfx1250) — matrix operand order repaired and checked by reading, not compiled, not run on hardware
//
#include <hip/hip_runtime.h>
#include <hip/hip_bf16.h>

typedef __attribute__((ext_vector_type(16))) __bf16 v16bf;
typedef __attribute__((ext_vector_type(8)))  float  v8f;

#define GN   8192
#define GE   262144
#define FIN_ 512
#define HID_ 512
#define LAT_ 256
#define FD_  512
#define NC_  16

#define BM 64
#define BN 64
#define BK 32

#define PA_   80    // LDS pitch (bytes): 64B row data + 16B pad (16B aligned)
#define PB_   80

// ===========================================================================
// Async double-buffered bf16 WMMA GEMM.
//   C[M,N] = epilogue(A[M,K] * B^T)
//   A: [M,K] bf16 row-major.  B: [N,K] bf16 row-major (i.e. operand already
//   transposed, so every LDS B tile is column-major -> contiguous fragments).
//   C fp32, fp32 accumulate.  EPI: 0 none, 1 +bias, 2 relu(+bias), 3 sigmoid.
//   Requires M%64==0, N%64==0, K%32==0.
// Tiles move global->LDS via global_load_async_to_lds_b128 (ASYNCcnt),
// double buffered: issue tile k+1, s_wait_asynccnt 4 -> tile k resident
// (async loads complete in order).
// ===========================================================================
template <int EPI>
__global__ __launch_bounds__(128) void wmma_gemm_async(
    const __bf16* __restrict__ A, const __bf16* __restrict__ B,
    const float* __restrict__ bias, float* __restrict__ C,
    int M, int N, int K)
{
    __shared__ unsigned char Araw[2][64 * PA_];    // 2 x 5120 B
    __shared__ unsigned char Braw[2][64 * PB_];    // 2 x 5120 B

    const int tid  = threadIdx.x;
    const int wave = tid >> 5;
    const int lane = tid & 31;
    const int wm   = wave >> 1;
    const int wn   = wave & 1;
    const int fr   = lane & 15;
    const int fh   = lane >> 4;

    const int m0 = blockIdx.y * BM;
    const int n0 = blockIdx.x * BN;

    // generic->LDS: low 32 bits of a flat pointer into LDS == LDS byte offset
    unsigned aBase[2] = { (unsigned)(uintptr_t)&Araw[0][0],
                          (unsigned)(uintptr_t)&Araw[1][0] };
    unsigned bBase[2] = { (unsigned)(uintptr_t)&Braw[0][0],
                          (unsigned)(uintptr_t)&Braw[1][0] };

    // Issue async copies for the K-chunk starting at k0 into buffer b.
    // A and B tiles are both 64 rows x 64B: 4 async instructions per wave.
    auto issue = [&](int k0, int b) {
        #pragma unroll
        for (int t = 0; t < 2; ++t) {
            int i = tid + t * 128;
            int r = i >> 2, q = i & 3;
            unsigned lds = aBase[b] + r * PA_ + q * 16;
            const __bf16* g = A + (long long)(m0 + r) * K + k0 + q * 8;
            asm volatile("global_load_async_to_lds_b128 %0, %1, off"
                         :: "v"(lds), "v"(g) : "memory");
        }
        #pragma unroll
        for (int t = 0; t < 2; ++t) {
            int i = tid + t * 128;
            int nn = i >> 2, q = i & 3;
            unsigned lds = bBase[b] + nn * PB_ + q * 16;
            const __bf16* g = B + (long long)(n0 + nn) * K + k0 + q * 8;
            asm volatile("global_load_async_to_lds_b128 %0, %1, off"
                         :: "v"(lds), "v"(g) : "memory");
        }
    };

    v8f acc[2][2] = {};
    const int nK = K / BK;

    issue(0, 0);
    for (int kk = 0; kk < nK; ++kk) {
        const int cur = kk & 1;
        if (kk + 1 < nK) {
            issue((kk + 1) * BK, cur ^ 1);
            asm volatile("s_wait_asynccnt 0x4" ::: "memory"); // tile kk resident
        } else {
            asm volatile("s_wait_asynccnt 0x0" ::: "memory");
        }
        __syncthreads();

        const unsigned char* Ab = Araw[cur];
        const unsigned char* Bb = Braw[cur];
        #pragma unroll
        for (int i = 0; i < 2; i++) {
            const __bf16* arow = (const __bf16*)(Ab + (wm * 32 + i * 16 + fr) * PA_);
            v16bf af;
            #pragma unroll
            for (int e = 0; e < 16; e++) {
                // ISA 16-bit A 16x32 layout: K = 8*h + e (+8 when e >= 8)
                int k = 8 * fh + e + ((e >= 8) ? 8 : 0);
                af[e] = arow[k];
            }
            #pragma unroll
            for (int j = 0; j < 2; j++) {
                // ISA 16-bit B 32x16 layout: col = lane%16, K = 8*h + e (+8 when e >= 8)
                const __bf16* brow =
                    (const __bf16*)(Bb + (wn * 32 + j * 16 + fr) * PB_);
                v16bf bv;
                #pragma unroll
                for (int e = 0; e < 16; e++) bv[e] = brow[8 * fh + e + ((e >= 8) ? 8 : 0)];
                acc[i][j] = __builtin_amdgcn_wmma_f32_16x16x32_bf16(
                    false, af, false, bv, (short)0, acc[i][j], false, false);
            }
        }
        __syncthreads();
    }

    // epilogue + store (C/D layout: row = d + 8*h, col = lane%16); N%64==0
    #pragma unroll
    for (int i = 0; i < 2; i++) {
        #pragma unroll
        for (int j = 0; j < 2; j++) {
            int col = n0 + wn * 32 + j * 16 + fr;
            float bval = (EPI == 1 || EPI == 2) ? bias[col] : 0.0f;
            #pragma unroll
            for (int d = 0; d < 8; d++) {
                int row = m0 + wm * 32 + i * 16 + d + 8 * fh;
                float v = acc[i][j][d];
                if (EPI == 1 || EPI == 2) v += bval;
                if (EPI == 2) v = fmaxf(v, 0.0f);
                if (EPI == 3) v = 1.0f / (1.0f + __expf(-v));
                C[(long long)row * N + col] = v;
            }
        }
    }
}

// ===========================================================================
// Synchronous fp32-input WMMA GEMM (kept only for the tiny N=16 head).
// B is [K,N] row-major here.
// ===========================================================================
template <int EPI>
__global__ __launch_bounds__(128) void wmma_gemm_f32(
    const float* __restrict__ A, const float* __restrict__ B,
    const float* __restrict__ bias, float* __restrict__ C,
    int M, int N, int K)
{
    __shared__ __bf16 As[BM][BK + 2];
    __shared__ __bf16 Bs[BN][BK + 2];

    const int tid  = threadIdx.x;
    const int wave = tid >> 5;
    const int lane = tid & 31;
    const int wm = wave >> 1, wn = wave & 1;
    const int fr = lane & 15, fh = lane >> 4;
    const int m0 = blockIdx.y * BM;
    const int n0 = blockIdx.x * BN;

    v8f acc[2][2] = {};

    for (int k0 = 0; k0 < K; k0 += BK) {
        for (int i = tid; i < BM * BK; i += 128) {
            int r = i >> 5, c = i & 31;
            As[r][c] = (__bf16)A[(long long)(m0 + r) * K + k0 + c];
        }
        for (int i = tid; i < BK * BN; i += 128) {
            int c = i & 63, r = i >> 6;
            int gn = n0 + c;
            float v = (gn < N) ? B[(long long)(k0 + r) * N + gn] : 0.0f;
            Bs[c][r] = (__bf16)v;
        }
        __syncthreads();
        #pragma unroll
        for (int i = 0; i < 2; i++) {
            v16bf af;
            #pragma unroll
            for (int e = 0; e < 16; e++) {
                int k = 8 * fh + e + ((e >= 8) ? 8 : 0);
                af[e] = As[wm * 32 + i * 16 + fr][k];
            }
            #pragma unroll
            for (int j = 0; j < 2; j++) {
                v16bf bv;
                #pragma unroll
                for (int e = 0; e < 16; e++)
                    bv[e] = Bs[wn * 32 + j * 16 + fr][8 * fh + e + ((e >= 8) ? 8 : 0)];
                acc[i][j] = __builtin_amdgcn_wmma_f32_16x16x32_bf16(
                    false, af, false, bv, (short)0, acc[i][j], false, false);
            }
        }
        __syncthreads();
    }
    #pragma unroll
    for (int i = 0; i < 2; i++)
        #pragma unroll
        for (int j = 0; j < 2; j++)
            #pragma unroll
            for (int d = 0; d < 8; d++) {
                int row = m0 + wm * 32 + i * 16 + d + 8 * fh;
                int col = n0 + wn * 32 + j * 16 + fr;
                if (col < N) {
                    float v = acc[i][j][d];
                    if (EPI == 1 || EPI == 2) v += bias[col];
                    if (EPI == 2) v = fmaxf(v, 0.0f);
                    if (EPI == 3) v = 1.0f / (1.0f + __expf(-v));
                    C[(long long)row * N + col] = v;
                }
            }
}

// ===========================================================================
// Graph + elementwise helpers
// ===========================================================================
__global__ void deg_kernel(const long long* __restrict__ dst,
                           float* __restrict__ deg, int E)
{
    int i = blockIdx.x * blockDim.x + threadIdx.x;
    if (i < E) unsafeAtomicAdd(&deg[dst[i]], 1.0f);
}

__global__ void dinv_kernel(const float* __restrict__ deg,
                            float* __restrict__ dinv, int n)
{
    int i = blockIdx.x * blockDim.x + threadIdx.x;
    if (i < n) dinv[i] = rsqrtf(deg[i] + 1.0f);   // +1 self loop
}

template <int DIM>
__global__ __launch_bounds__(256) void aggregate_kernel(
    const float* __restrict__ feat, float* __restrict__ agg,
    const long long* __restrict__ srcI, const long long* __restrict__ dstI,
    const float* __restrict__ dinv, int E, int Nn)
{
    int gw   = (blockIdx.x * blockDim.x + threadIdx.x) >> 5;
    int lane = threadIdx.x & 31;
    if (gw >= E + Nn) return;
    long long s, d;
    if (gw < E) { s = srcI[gw]; d = dstI[gw]; }
    else        { s = d = gw - E; }
    float nrm = dinv[s] * dinv[d];
    const float* in = feat + s * DIM;
    float*      out = agg  + d * DIM;
    #pragma unroll
    for (int c = lane; c < DIM; c += 32)
        unsafeAtomicAdd(&out[c], in[c] * nrm);
}

__global__ void f32_to_bf16_kernel(const float* __restrict__ in,
                                   __bf16* __restrict__ out, long long n)
{
    long long i = (long long)blockIdx.x * blockDim.x + threadIdx.x;
    if (i < n) out[i] = (__bf16)in[i];
}

// W [K,N] fp32 row-major -> Wt [N,K] bf16 row-major (coalesced writes)
__global__ void transpose_to_bf16_kernel(const float* __restrict__ in,
                                         __bf16* __restrict__ out, int K, int N)
{
    long long i = (long long)blockIdx.x * blockDim.x + threadIdx.x;
    if (i < (long long)K * N) {
        int n = (int)(i / K), k = (int)(i % K);
        out[i] = (__bf16)in[(long long)k * N + n];
    }
}

__global__ void bias_act_kernel(float* __restrict__ h, const float* __restrict__ b,
                                long long total, int dim, int relu)
{
    long long i = (long long)blockIdx.x * blockDim.x + threadIdx.x;
    if (i < total) {
        float v = h[i] + b[i % dim];
        if (relu) v = fmaxf(v, 0.0f);
        h[i] = v;
    }
}

__global__ void z_kernel(const float* __restrict__ mu, const float* __restrict__ lv,
                         const float* __restrict__ eps, float* __restrict__ z,
                         long long total)
{
    long long i = (long long)blockIdx.x * blockDim.x + threadIdx.x;
    if (i < total) z[i] = mu[i] + eps[i] * __expf(0.5f * lv[i]);
}

__global__ void add_kernel(const float* __restrict__ a, const float* __restrict__ b,
                           float* __restrict__ c, long long total)
{
    long long i = (long long)blockIdx.x * blockDim.x + threadIdx.x;
    if (i < total) c[i] = a[i] + b[i];
}

static inline int div_up_i(long long a, long long b) { return (int)((a + b - 1) / b); }

// ===========================================================================
extern "C" void kernel_launch(void* const* d_in, const int* in_sizes, int n_in,
                              void* d_out, int out_size, void* d_ws, size_t ws_size,
                              hipStream_t stream)
{
    (void)in_sizes; (void)n_in; (void)out_size; (void)ws_size;

    const float*     x    = (const float*)d_in[0];
    const long long* ei   = (const long long*)d_in[1];
    const float*     W1   = (const float*)d_in[2];
    const float*     b1   = (const float*)d_in[3];
    const float*     Wmu  = (const float*)d_in[4];
    const float*     bmu  = (const float*)d_in[5];
    const float*     Wlv  = (const float*)d_in[6];
    const float*     blv  = (const float*)d_in[7];
    const float*     Wc   = (const float*)d_in[8];
    const float*     bc   = (const float*)d_in[9];
    const float*     Wf   = (const float*)d_in[10];
    const float*     bf_  = (const float*)d_in[11];
    const float*     Wl1  = (const float*)d_in[12];
    const float*     bl1  = (const float*)d_in[13];
    const float*     Wl2  = (const float*)d_in[14];
    const float*     bl2  = (const float*)d_in[15];
    const float*     eps  = (const float*)d_in[16];

    float* out  = (float*)d_out;
    float* o_mu = out;
    float* o_lv = out + (long long)GN * LAT_;
    float* o_z  = out + 2LL * GN * LAT_;
    float* o_A  = out + 3LL * GN * LAT_;
    float* o_X  = o_A + (long long)GN * GN;
    float* o_Y  = o_X + (long long)GN * FD_;

    // ---- workspace layout ----
    float* ws   = (float*)d_ws;
    float* deg  = ws;
    float* dinv = deg  + GN;
    float* buf1 = dinv + GN;                      // GN*HID  fp32 (t1, later hl)
    float* buf2 = buf1 + (long long)GN * HID_;    // GN*HID  fp32 (agg1 -> h1)
    float* buf3 = buf2 + (long long)GN * HID_;    // GN*LAT  fp32 (mu_t, later xc)
    float* buf4 = buf3 + (long long)GN * LAT_;    // GN*LAT  fp32 (lv_t, later h)
    __bf16* xb   = (__bf16*)(buf4 + (long long)GN * LAT_);  // GN*FIN
    __bf16* h1b  = xb  + (long long)GN * FIN_;    // GN*HID
    __bf16* zb   = h1b + (long long)GN * HID_;    // GN*LAT
    __bf16* hb   = zb  + (long long)GN * LAT_;    // GN*LAT
    __bf16* W1t  = hb  + (long long)GN * LAT_;    // [HID,FIN]
    __bf16* Wmut = W1t  + (long long)FIN_ * HID_; // [LAT,HID]
    __bf16* Wlvt = Wmut + (long long)HID_ * LAT_; // [LAT,HID]
    __bf16* Wct  = Wlvt + (long long)HID_ * LAT_; // [LAT,FIN]
    __bf16* Wft  = Wct  + (long long)FIN_ * LAT_; // [FD,LAT]
    __bf16* Wl1t = Wft  + (long long)LAT_ * FD_;  // [HID,LAT]

    const long long* srcI = ei;
    const long long* dstI = ei + GE;

    const long long nLat = (long long)GN * LAT_;
    const long long nHid = (long long)GN * HID_;
    const int nwork = GE + GN;
    const int aggBlocks = div_up_i((long long)nwork * 32, 256);

    // ---- operand conversions: x -> bf16, weights -> transposed bf16 ----
    f32_to_bf16_kernel<<<div_up_i((long long)GN * FIN_, 256), 256, 0, stream>>>(x, xb, (long long)GN * FIN_);
    transpose_to_bf16_kernel<<<div_up_i((long long)FIN_ * HID_, 256), 256, 0, stream>>>(W1, W1t, FIN_, HID_);
    transpose_to_bf16_kernel<<<div_up_i((long long)HID_ * LAT_, 256), 256, 0, stream>>>(Wmu, Wmut, HID_, LAT_);
    transpose_to_bf16_kernel<<<div_up_i((long long)HID_ * LAT_, 256), 256, 0, stream>>>(Wlv, Wlvt, HID_, LAT_);
    transpose_to_bf16_kernel<<<div_up_i((long long)FIN_ * LAT_, 256), 256, 0, stream>>>(Wc, Wct, FIN_, LAT_);
    transpose_to_bf16_kernel<<<div_up_i((long long)LAT_ * FD_, 256), 256, 0, stream>>>(Wf, Wft, LAT_, FD_);
    transpose_to_bf16_kernel<<<div_up_i((long long)LAT_ * HID_, 256), 256, 0, stream>>>(Wl1, Wl1t, LAT_, HID_);

    // ---- degree / normalization ----
    hipMemsetAsync(deg, 0, GN * sizeof(float), stream);
    deg_kernel<<<div_up_i(GE, 256), 256, 0, stream>>>(dstI, deg, GE);
    dinv_kernel<<<div_up_i(GN, 256), 256, 0, stream>>>(deg, dinv, GN);

    // ---- layer 1: t1 = x @ W1 ; h1 = relu(scatter(t1) + b1) ----
    dim3 g1(HID_ / BN, GN / BM);
    wmma_gemm_async<0><<<g1, 128, 0, stream>>>(xb, W1t, nullptr, buf1, GN, HID_, FIN_);
    hipMemsetAsync(buf2, 0, (size_t)nHid * sizeof(float), stream);
    aggregate_kernel<HID_><<<aggBlocks, 256, 0, stream>>>(buf1, buf2, srcI, dstI, dinv, GE, GN);
    bias_act_kernel<<<div_up_i(nHid, 256), 256, 0, stream>>>(buf2, b1, nHid, HID_, 1);
    f32_to_bf16_kernel<<<div_up_i(nHid, 256), 256, 0, stream>>>(buf2, h1b, nHid);

    // ---- mu / logvar GCN heads ----
    dim3 g2(LAT_ / BN, GN / BM);
    wmma_gemm_async<0><<<g2, 128, 0, stream>>>(h1b, Wmut, nullptr, buf3, GN, LAT_, HID_);
    wmma_gemm_async<0><<<g2, 128, 0, stream>>>(h1b, Wlvt, nullptr, buf4, GN, LAT_, HID_);
    hipMemsetAsync(o_mu, 0, (size_t)nLat * sizeof(float), stream);
    hipMemsetAsync(o_lv, 0, (size_t)nLat * sizeof(float), stream);
    aggregate_kernel<LAT_><<<aggBlocks, 256, 0, stream>>>(buf3, o_mu, srcI, dstI, dinv, GE, GN);
    aggregate_kernel<LAT_><<<aggBlocks, 256, 0, stream>>>(buf4, o_lv, srcI, dstI, dinv, GE, GN);
    bias_act_kernel<<<div_up_i(nLat, 256), 256, 0, stream>>>(o_mu, bmu, nLat, LAT_, 0);
    bias_act_kernel<<<div_up_i(nLat, 256), 256, 0, stream>>>(o_lv, blv, nLat, LAT_, 0);

    // ---- z = mu + eps * exp(0.5*logvar) ----
    z_kernel<<<div_up_i(nLat, 256), 256, 0, stream>>>(o_mu, o_lv, eps, o_z, nLat);
    f32_to_bf16_kernel<<<div_up_i(nLat, 256), 256, 0, stream>>>(o_z, zb, nLat);

    // ---- h = z + relu(x @ Wc + bc) ----
    wmma_gemm_async<2><<<g2, 128, 0, stream>>>(xb, Wct, bc, buf3, GN, LAT_, FIN_);
    add_kernel<<<div_up_i(nLat, 256), 256, 0, stream>>>(o_z, buf3, buf4, nLat);
    f32_to_bf16_kernel<<<div_up_i(nLat, 256), 256, 0, stream>>>(buf4, hb, nLat);

    // ---- A_pred = sigmoid(h @ h^T)  (dominant GEMM: 8192x8192x256) ----
    dim3 gA(GN / BN, GN / BM);
    wmma_gemm_async<3><<<gA, 128, 0, stream>>>(hb, hb, nullptr, o_A, GN, GN, LAT_);

    // ---- X_pred = z @ Wf + bf ----
    dim3 gX(FD_ / BN, GN / BM);
    wmma_gemm_async<1><<<gX, 128, 0, stream>>>(zb, Wft, bf_, o_X, GN, FD_, LAT_);

    // ---- Y_logits = relu(z @ Wl1 + bl1) @ Wl2 + bl2 ----
    dim3 gH(HID_ / BN, GN / BM);
    wmma_gemm_async<2><<<gH, 128, 0, stream>>>(zb, Wl1t, bl1, buf1, GN, HID_, LAT_);
    dim3 gY(1, GN / BM);
    wmma_gemm_f32<1><<<gY, 128, 0, stream>>>(buf1, Wl2, bl2, o_Y, GN, NC_, HID_);
}
